// SuperGlue_17987323036178
// MI455X (gfx1250) — matrix operand order repaired and checked by reading, not compiled, not run on hardware
//
#include <hip/hip_runtime.h>
#include <hip/hip_bf16.h>
#include <math.h>

typedef _Float16 h16;
typedef __attribute__((ext_vector_type(16))) _Float16 v16h;
typedef __attribute__((ext_vector_type(8)))  _Float16 v8h;
typedef __attribute__((ext_vector_type(8)))  float    v8f;

// ---------------------------------------------------------------------------
// WMMA GEMM:  Y[r,o] = (sum_k X[r,k]*W[o,k] + bias[o]) * scale, opt relu.
// W: [O, Kp] f16 row-major, X: [rows, Kp] f16 row-major.
// Output written at Y[(r)*ldY + colOff + o] (f32 and/or f16).
//
// Register-blocked: one wave computes a 32(o) x 64(r) tile = 2x4 WMMA subtiles,
// so each K-step does 8 v_wmma_f32_16x16x32_f16 against 12 b128 loads
// (A frags reused across 4 B frags, B frags across 2 A frags).
//  A frag (16x32, 16-bit):  lane=16g+m holds row m; half h -> K=16g+h
//                           (two 16B loads)
//  B frag (32x16, 16-bit):  lane=16g+n holds col n; half h -> K=16g+h (one 32B load)
//  C/D  : lane=16g+n, vgpr r -> (o = 8g+r, col n)  -> 8 contiguous outputs/lane
// Grid: (O/32, rows/256), block (32,4): wave w -> rT = by*4+w.
// ---------------------------------------------------------------------------
__global__ __launch_bounds__(128) void sg_gemm_wmma(
    const h16* __restrict__ W, const h16* __restrict__ X,
    const float* __restrict__ bias,
    float* __restrict__ Yf, h16* __restrict__ Yh,
    int Kp, int ldY, int colOff, float scale, int relu)
{
  const int lane = threadIdx.x;
  const int g = lane >> 4, l = lane & 15;
  const int oT = blockIdx.x;                     // 32 output channels
  const int rT = blockIdx.y * 4 + threadIdx.y;   // 64 rows
  const h16* wrow0 = W + (size_t)(oT * 32 + l) * Kp;
  const h16* wrow1 = wrow0 + (size_t)16 * Kp;
  const h16* xrow0 = X + (size_t)(rT * 64 + l) * Kp;
  const h16* xrow1 = xrow0 + (size_t)16 * Kp;
  const h16* xrow2 = xrow0 + (size_t)32 * Kp;
  const h16* xrow3 = xrow0 + (size_t)48 * Kp;
  v8f c[2][4];
#pragma unroll
  for (int i = 0; i < 2; ++i)
#pragma unroll
    for (int j = 0; j < 4; ++j)
      c[i][j] = (v8f){0.f, 0.f, 0.f, 0.f, 0.f, 0.f, 0.f, 0.f};

  for (int k0 = 0; k0 < Kp; k0 += 32) {
    union { v16h v; v8h h[2]; } a0, a1, b0, b1, b2, b3;
    a0.h[0] = *(const v8h*)(wrow0 + k0 + 16 * g);
    a0.h[1] = *(const v8h*)(wrow0 + k0 + 16 * g + 8);
    a1.h[0] = *(const v8h*)(wrow1 + k0 + 16 * g);
    a1.h[1] = *(const v8h*)(wrow1 + k0 + 16 * g + 8);
    b0.v    = *(const v16h*)(xrow0 + k0 + 16 * g);
    b1.v    = *(const v16h*)(xrow1 + k0 + 16 * g);
    b2.v    = *(const v16h*)(xrow2 + k0 + 16 * g);
    b3.v    = *(const v16h*)(xrow3 + k0 + 16 * g);
    c[0][0] = __builtin_amdgcn_wmma_f32_16x16x32_f16(false, a0.v, false, b0.v, (short)0, c[0][0], false, false);
    c[0][1] = __builtin_amdgcn_wmma_f32_16x16x32_f16(false, a0.v, false, b1.v, (short)0, c[0][1], false, false);
    c[0][2] = __builtin_amdgcn_wmma_f32_16x16x32_f16(false, a0.v, false, b2.v, (short)0, c[0][2], false, false);
    c[0][3] = __builtin_amdgcn_wmma_f32_16x16x32_f16(false, a0.v, false, b3.v, (short)0, c[0][3], false, false);
    c[1][0] = __builtin_amdgcn_wmma_f32_16x16x32_f16(false, a1.v, false, b0.v, (short)0, c[1][0], false, false);
    c[1][1] = __builtin_amdgcn_wmma_f32_16x16x32_f16(false, a1.v, false, b1.v, (short)0, c[1][1], false, false);
    c[1][2] = __builtin_amdgcn_wmma_f32_16x16x32_f16(false, a1.v, false, b2.v, (short)0, c[1][2], false, false);
    c[1][3] = __builtin_amdgcn_wmma_f32_16x16x32_f16(false, a1.v, false, b3.v, (short)0, c[1][3], false, false);
  }

#pragma unroll
  for (int i = 0; i < 2; ++i) {
    const int ocol = oT * 32 + i * 16 + 8 * g;
    float bv[8];
#pragma unroll
    for (int r = 0; r < 8; ++r) bv[r] = bias ? bias[ocol + r] : 0.f;
#pragma unroll
    for (int j = 0; j < 4; ++j) {
      v8f cc = c[i][j];
#pragma unroll
      for (int r = 0; r < 8; ++r) {
        cc[r] = (cc[r] + bv[r]) * scale;
        if (relu) cc[r] = fmaxf(cc[r], 0.f);
      }
      const size_t base = (size_t)(rT * 64 + 16 * j + l) * ldY + colOff + ocol;
      if (Yf) {
#pragma unroll
        for (int r = 0; r < 8; ++r) Yf[base + r] = cc[r];
      }
      if (Yh) {
        v8h o;
#pragma unroll
        for (int r = 0; r < 8; ++r) o[r] = (h16)cc[r];
        *(v8h*)(Yh + base) = o;
      }
    }
  }
}

// f32 weight [O,C] -> f16 [O,Cp] with zero padding of K dim
__global__ void sg_cvt_w(const float* __restrict__ W, h16* __restrict__ W16,
                         int C, int Cp, int total)
{
  int t = blockIdx.x * 256 + threadIdx.x;
  if (t >= total) return;
  int o = t / Cp, c = t % Cp;
  W16[t] = (c < C) ? (h16)W[(size_t)o * C + c] : (h16)0.f;
}

// keypoint encoder input: [2048 rows, 32] f16, channels {k2x, k2y, score, 0...}
__global__ void sg_build_enc(const float* __restrict__ kp, const float* __restrict__ sc,
                             const int* __restrict__ imsz, h16* __restrict__ E)
{
  int t = blockIdx.x * 256 + threadIdx.x;
  if (t >= 2048 * 32) return;
  int r = t >> 5, c = t & 31;
  int b = r >> 10, n = r & 1023;
  float hh = (float)imsz[0], ww = (float)imsz[1];       // image_size[0] = (h, w)
  float cx = ww * 0.5f, cy = hh * 0.5f;
  float scal = fmaxf(ww, hh) * 0.7f;
  float v = 0.f;
  if (c == 0)      v = (kp[((size_t)b * 1024 + n) * 2 + 0] - cx) / scal;
  else if (c == 1) v = (kp[((size_t)b * 1024 + n) * 2 + 1] - cy) / scal;
  else if (c == 2) v = sc[(size_t)b * 1024 + n];
  E[t] = (h16)v;
}

// BatchNorm over rows (per channel) + ReLU, f32 in -> f16 out. grid.x = C.
__global__ void sg_bn_relu(const float* __restrict__ Y, const float* __restrict__ gam,
                           const float* __restrict__ bet, h16* __restrict__ out,
                           int rows, int C)
{
  int c = blockIdx.x, tid = threadIdx.x;
  __shared__ float s1[256], s2[256];
  float sum = 0.f, sq = 0.f;
  for (int r = tid; r < rows; r += 256) {
    float v = Y[(size_t)r * C + c];
    sum += v; sq += v * v;
  }
  s1[tid] = sum; s2[tid] = sq; __syncthreads();
  for (int s = 128; s > 0; s >>= 1) {
    if (tid < s) { s1[tid] += s1[tid + s]; s2[tid] += s2[tid + s]; }
    __syncthreads();
  }
  float mean = s1[0] / (float)rows;
  float var  = s2[0] / (float)rows - mean * mean;
  float inv  = rsqrtf(var + 1e-5f) * gam[c];
  float beta = bet[c];
  for (int r = tid; r < rows; r += 256) {
    float v = (Y[(size_t)r * C + c] - mean) * inv + beta;
    out[(size_t)r * C + c] = (h16)fmaxf(v, 0.f);
  }
}

// initialize master activation: dM/X16 row = b*1024+n (+rowOff), from desc[b,c,n] (+add)
__global__ void sg_init_master(const float* __restrict__ desc, const float* __restrict__ add,
                               float* __restrict__ dM, h16* __restrict__ X16, int rowOff)
{
  int t = blockIdx.x * 256 + threadIdx.x;
  if (t >= 2048 * 256) return;
  int r = t >> 8, c = t & 255;
  int b = r >> 10, n = r & 1023;
  float v = desc[((size_t)b * 256 + c) * 1024 + n];
  if (add) v += add[t];
  size_t o = (size_t)(rowOff + r) * 256 + c;
  dM[o] = v; X16[o] = (h16)v;
}

// head gather: dst[s][n][64] = src[row(s,n)][d*4+h]   (s = (w*2+b)*4+h)
__global__ void sg_gather_nk(const h16* __restrict__ src, h16* __restrict__ dst)
{
  int t = blockIdx.x * 256 + threadIdx.x;           // 16*1024*64
  int s = t >> 16, rem = t & 65535;
  int n = rem >> 6, d = rem & 63;
  int w = s >> 3, b = (s >> 2) & 1, h = s & 3;
  int row = (w * 2 + b) * 1024 + n;
  dst[t] = src[(size_t)row * 256 + d * 4 + h];
}

// V transposed gather: dst[s][d][1024] = src[row(s,m)][d*4+h]
__global__ void sg_gather_vT(const h16* __restrict__ src, h16* __restrict__ dst)
{
  int t = blockIdx.x * 256 + threadIdx.x;
  int s = t >> 16, rem = t & 65535;
  int d = rem >> 10, m = rem & 1023;
  int w = s >> 3, b = (s >> 2) & 1, h = s & 3;
  int row = (w * 2 + b) * 1024 + m;
  dst[t] = src[(size_t)row * 256 + d * 4 + h];
}

// merge heads: Om[row][d*4+h] = Oh[s][n][d]
__global__ void sg_scatter_heads(const h16* __restrict__ Oh, h16* __restrict__ Om)
{
  int t = blockIdx.x * 256 + threadIdx.x;
  int s = t >> 16, rem = t & 65535;
  int n = rem >> 6, d = rem & 63;
  int w = s >> 3, b = (s >> 2) & 1, h = s & 3;
  int row = (w * 2 + b) * 1024 + n;
  Om[(size_t)row * 256 + d * 4 + h] = Oh[t];
}

// row softmax with scale, f32 -> f16
__global__ void sg_softmax_row(const float* __restrict__ S, h16* __restrict__ P,
                               int n, float scale)
{
  const float* row = S + (size_t)blockIdx.x * n;
  h16* prow = P + (size_t)blockIdx.x * n;
  __shared__ float red[256];
  int tid = threadIdx.x;
  float mx = -3.4e38f;
  for (int j = tid; j < n; j += 256) mx = fmaxf(mx, row[j] * scale);
  red[tid] = mx; __syncthreads();
  for (int s = 128; s > 0; s >>= 1) {
    if (tid < s) red[tid] = fmaxf(red[tid], red[tid + s]);
    __syncthreads();
  }
  mx = red[0]; __syncthreads();
  float sum = 0.f;
  for (int j = tid; j < n; j += 256) sum += __expf(row[j] * scale - mx);
  red[tid] = sum; __syncthreads();
  for (int s = 128; s > 0; s >>= 1) {
    if (tid < s) red[tid] += red[tid + s];
    __syncthreads();
  }
  float inv = 1.f / red[0];
  for (int j = tid; j < n; j += 256) prow[j] = (h16)(__expf(row[j] * scale - mx) * inv);
}

// cat[:, 0:256] = X16
__global__ void sg_copy_cols(const h16* __restrict__ X, h16* __restrict__ cat)
{
  int t = blockIdx.x * 256 + threadIdx.x;
  if (t >= 4096 * 256) return;
  int r = t >> 8, c = t & 255;
  cat[(size_t)r * 512 + c] = X[t];
}

// d += delta ; X16 = (f16)d
__global__ void sg_residual(float* __restrict__ dM, const float* __restrict__ delta,
                            h16* __restrict__ X16, int n)
{
  int t = blockIdx.x * 256 + threadIdx.x;
  if (t >= n) return;
  float v = dM[t] + delta[t];
  dM[t] = v; X16[t] = (h16)v;
}

// row L2 normalize (256 cols), f32 -> f16
__global__ void sg_norm_rows(const float* __restrict__ Mf, h16* __restrict__ M16)
{
  int r = blockIdx.x, tid = threadIdx.x;
  __shared__ float red[256];
  float v = Mf[(size_t)r * 256 + tid];
  red[tid] = v * v; __syncthreads();
  for (int s = 128; s > 0; s >>= 1) {
    if (tid < s) red[tid] += red[tid + s];
    __syncthreads();
  }
  float nrm = fmaxf(sqrtf(red[0]), 1e-12f);
  M16[(size_t)r * 256 + tid] = (h16)(v / nrm);
}

// couplings: [b][1025][1025], bordered with alpha
__global__ void sg_build_cp(const float* __restrict__ S, const float* __restrict__ alpha,
                            float* __restrict__ cp, long total)
{
  long t = (long)blockIdx.x * 256 + threadIdx.x;
  if (t >= total) return;
  long per = 1025L * 1025L;
  int b = (int)(t / per);
  int rem = (int)(t % per);
  int i = rem / 1025, j = rem % 1025;
  cp[t] = (i < 1024 && j < 1024) ? S[((size_t)b * 1024 + i) * 1024 + j] : alpha[0];
}

__global__ void sg_fill0(float* __restrict__ p, int n)
{
  int t = blockIdx.x * 256 + threadIdx.x;
  if (t < n) p[t] = 0.f;
}

// u[b][i] = log_mu[i] - LSE_j(cp[b][i][j] + v[b][j])
__global__ void sg_sink_u(const float* __restrict__ cp, const float* __restrict__ v,
                          float* __restrict__ u, float norm, float lmu_last)
{
  int b = blockIdx.y, i = blockIdx.x, tid = threadIdx.x;
  const float* row = cp + ((size_t)b * 1025 + i) * 1025;
  const float* vb = v + b * 1025;
  __shared__ float red[256];
  float mx = -3.4e38f;
  for (int j = tid; j < 1025; j += 256) mx = fmaxf(mx, row[j] + vb[j]);
  red[tid] = mx; __syncthreads();
  for (int s = 128; s > 0; s >>= 1) {
    if (tid < s) red[tid] = fmaxf(red[tid], red[tid + s]);
    __syncthreads();
  }
  mx = red[0]; __syncthreads();
  float sum = 0.f;
  for (int j = tid; j < 1025; j += 256) sum += __expf(row[j] + vb[j] - mx);
  red[tid] = sum; __syncthreads();
  for (int s = 128; s > 0; s >>= 1) {
    if (tid < s) red[tid] += red[tid + s];
    __syncthreads();
  }
  if (tid == 0) {
    float lse = mx + __logf(red[0]);
    u[b * 1025 + i] = ((i < 1024) ? norm : lmu_last) - lse;
  }
}

// v[b][j] = log_nu[j] - LSE_i(cp[b][i][j] + u[b][i])
__global__ void sg_sink_v(const float* __restrict__ cp, const float* __restrict__ u,
                          float* __restrict__ v, float norm, float lnu_last)
{
  int b = blockIdx.y, j = blockIdx.x, tid = threadIdx.x;
  const float* base = cp + (size_t)b * 1025 * 1025 + j;
  const float* ub = u + b * 1025;
  __shared__ float red[256];
  float mx = -3.4e38f;
  for (int i = tid; i < 1025; i += 256) mx = fmaxf(mx, base[(size_t)i * 1025] + ub[i]);
  red[tid] = mx; __syncthreads();
  for (int s = 128; s > 0; s >>= 1) {
    if (tid < s) red[tid] = fmaxf(red[tid], red[tid + s]);
    __syncthreads();
  }
  mx = red[0]; __syncthreads();
  float sum = 0.f;
  for (int i = tid; i < 1025; i += 256) sum += __expf(base[(size_t)i * 1025] + ub[i] - mx);
  red[tid] = sum; __syncthreads();
  for (int s = 128; s > 0; s >>= 1) {
    if (tid < s) red[tid] += red[tid + s];
    __syncthreads();
  }
  if (tid == 0) {
    float lse = mx + __logf(red[0]);
    v[b * 1025 + j] = ((j < 1024) ? norm : lnu_last) - lse;
  }
}

// conf = exp(cp + u + v - norm), in place
__global__ void sg_conf(float* __restrict__ cp, const float* __restrict__ u,
                        const float* __restrict__ v, float norm, long total)
{
  long t = (long)blockIdx.x * 256 + threadIdx.x;
  if (t >= total) return;
  long per = 1025L * 1025L;
  int b = (int)(t / per);
  int rem = (int)(t % per);
  int i = rem / 1025, j = rem % 1025;
  cp[t] = __expf(cp[t] + u[b * 1025 + i] + v[b * 1025 + j] - norm);
}

__global__ void sg_rowmax(const float* __restrict__ conf, float* __restrict__ mx,
                          int* __restrict__ idx)
{
  int b = blockIdx.y, i = blockIdx.x, tid = threadIdx.x;
  const float* row = conf + ((size_t)b * 1025 + i) * 1025;
  __shared__ float rv[256]; __shared__ int ri[256];
  float bv = -3.4e38f; int bi = 0;
  for (int j = tid; j < 1024; j += 256) {
    float x = row[j];
    if (x > bv) { bv = x; bi = j; }
  }
  rv[tid] = bv; ri[tid] = bi; __syncthreads();
  for (int s = 128; s > 0; s >>= 1) {
    if (tid < s) {
      if (rv[tid + s] > rv[tid] || (rv[tid + s] == rv[tid] && ri[tid + s] < ri[tid])) {
        rv[tid] = rv[tid + s]; ri[tid] = ri[tid + s];
      }
    }
    __syncthreads();
  }
  if (tid == 0) { mx[b * 1024 + i] = rv[0]; idx[b * 1024 + i] = ri[0]; }
}

__global__ void sg_colmax(const float* __restrict__ conf, float* __restrict__ mx,
                          int* __restrict__ idx)
{
  int b = blockIdx.y, j = blockIdx.x, tid = threadIdx.x;
  const float* base = conf + (size_t)b * 1025 * 1025 + j;
  __shared__ float rv[256]; __shared__ int ri[256];
  float bv = -3.4e38f; int bi = 0;
  for (int i = tid; i < 1024; i += 256) {
    float x = base[(size_t)i * 1025];
    if (x > bv) { bv = x; bi = i; }
  }
  rv[tid] = bv; ri[tid] = bi; __syncthreads();
  for (int s = 128; s > 0; s >>= 1) {
    if (tid < s) {
      if (rv[tid + s] > rv[tid] || (rv[tid + s] == rv[tid] && ri[tid + s] < ri[tid])) {
        rv[tid] = rv[tid + s]; ri[tid] = ri[tid + s];
      }
    }
    __syncthreads();
  }
  if (tid == 0) { mx[b * 1024 + j] = rv[0]; idx[b * 1024 + j] = ri[0]; }
}

// outputs: [i0 int x2048][i1 int x2048][ms0 f32 x2048][ms1 f32 x2048]
__global__ void sg_finalize(const float* __restrict__ mx0, const int* __restrict__ id0,
                            const float* __restrict__ mx1, const int* __restrict__ id1,
                            void* __restrict__ outv)
{
  int t = blockIdx.x * 256 + threadIdx.x;
  if (t >= 2048) return;
  int b = t >> 10, n = t & 1023;
  int* oi0 = (int*)outv;
  int* oi1 = oi0 + 2048;
  float* oms0 = (float*)outv + 4096;
  float* oms1 = (float*)outv + 6144;
  const int* i0b = id0 + b * 1024; const int* i1b = id1 + b * 1024;
  const float* m0b = mx0 + b * 1024;
  // row side
  int j = i0b[n];
  bool mut0 = (i1b[j] == n);
  float ms0 = mut0 ? m0b[n] : 0.f;
  bool val0 = mut0 && (ms0 > 0.2f);
  oi0[t] = val0 ? j : -1;
  oms0[t] = ms0;
  // col side (m = n)
  int n2 = i1b[n];
  bool mut1 = (i0b[n2] == n);
  int j2 = i0b[n2];
  bool mut0b = (i1b[j2] == n2);
  float ms0b = mut0b ? m0b[n2] : 0.f;
  bool val0b = mut0b && (ms0b > 0.2f);
  oi1[t] = (mut1 && val0b) ? n2 : -1;
  oms1[t] = mut1 ? ms0b : 0.f;
}

// ---------------------------------------------------------------------------
extern "C" void kernel_launch(void* const* d_in, const int* in_sizes, int n_in,
                              void* d_out, int out_size, void* d_ws, size_t ws_size,
                              hipStream_t stream)
{
  (void)in_sizes; (void)n_in; (void)out_size; (void)ws_size;
  const float* desc2d = (const float*)d_in[0];
  const float* desc3d = (const float*)d_in[1];
  const float* kpts2d = (const float*)d_in[2];
  const float* sc2d   = (const float*)d_in[4];
  const int*   imsz   = (const int*)d_in[6];
  const float* fb     = (const float*)d_in[7];
  const float* fw     = (const float*)d_in[8];
  // gnn leaves: base 9, stride 14, sorted keys:
  // bk bm bq bv m1b m1bt m1g m1w m2b m2w wk wm wq wv
  const int P_GNN = 9, GS = 14;
  const int P_KENC = 9 + 18 * 14;       // 261
  const float* binsc = (const float*)d_in[P_KENC + 18];   // 279

  size_t off = 0;
  auto alloc = [&](size_t bytes) -> void* {
    off = (off + 255) & ~(size_t)255;
    void* p = (void*)((char*)d_ws + off);
    off += bytes;
    return p;
  };

  // f16 weights
  h16 *Wq[18], *Wk[18], *Wv[18], *Wm[18], *W1[18], *W2[18];
  for (int L = 0; L < 18; ++L) {
    Wq[L] = (h16*)alloc(256 * 256 * 2); Wk[L] = (h16*)alloc(256 * 256 * 2);
    Wv[L] = (h16*)alloc(256 * 256 * 2); Wm[L] = (h16*)alloc(256 * 256 * 2);
    W1[L] = (h16*)alloc(512 * 512 * 2); W2[L] = (h16*)alloc(256 * 512 * 2);
  }
  const int kO[5]  = {32, 64, 128, 256, 256};
  const int kC[5]  = {3, 32, 64, 128, 256};
  const int kCp[5] = {32, 32, 64, 128, 256};
  h16* KW[5];
  for (int i = 0; i < 5; ++i) KW[i] = (h16*)alloc((size_t)kO[i] * kCp[i] * 2);
  h16* FW16 = (h16*)alloc(256 * 256 * 2);

  // activations / scratch
  h16*  E16  = (h16*)alloc(2048 * 32 * 2);
  float* kF  = (float*)alloc((size_t)2048 * 256 * 4);
  h16*  kA   = (h16*)alloc((size_t)2048 * 256 * 2);
  float* kOut = (float*)alloc((size_t)2048 * 256 * 4);
  float* dM  = (float*)alloc((size_t)4096 * 256 * 4);
  h16*  X16  = (h16*)alloc((size_t)4096 * 256 * 2);
  h16*  Q16  = (h16*)alloc((size_t)4096 * 256 * 2);
  h16*  K16  = (h16*)alloc((size_t)4096 * 256 * 2);
  h16*  V16  = (h16*)alloc((size_t)4096 * 256 * 2);
  h16*  Qh   = (h16*)alloc((size_t)16 * 1024 * 64 * 2);
  h16*  Kh   = (h16*)alloc((size_t)16 * 1024 * 64 * 2);
  h16*  VhT  = (h16*)alloc((size_t)16 * 1024 * 64 * 2);
  h16*  Oh   = (h16*)alloc((size_t)16 * 1024 * 64 * 2);
  float* Sc1 = (float*)alloc((size_t)1024 * 1024 * 4);
  h16*  P16  = (h16*)alloc((size_t)1024 * 1024 * 2);
  h16*  Om16 = (h16*)alloc((size_t)4096 * 256 * 2);
  h16*  cat16 = (h16*)alloc((size_t)4096 * 512 * 2);
  float* m1f = (float*)alloc((size_t)4096 * 512 * 4);
  h16*  m1a  = (h16*)alloc((size_t)4096 * 512 * 2);
  float* m2f = (float*)alloc((size_t)4096 * 256 * 4);
  float* Mf  = (float*)alloc((size_t)4096 * 256 * 4);
  h16*  M16  = (h16*)alloc((size_t)4096 * 256 * 2);
  float* Smatch = (float*)alloc((size_t)2 * 1024 * 1024 * 4);
  float* cp  = (float*)alloc((size_t)2 * 1025 * 1025 * 4);
  float* u   = (float*)alloc(2 * 1025 * 4);
  float* v   = (float*)alloc(2 * 1025 * 4);
  float* mx0 = (float*)alloc(2 * 1024 * 4);
  int*   id0 = (int*)alloc(2 * 1024 * 4);
  float* mx1 = (float*)alloc(2 * 1024 * 4);
  int*   id1 = (int*)alloc(2 * 1024 * 4);

  auto cvt = [&](const float* W, h16* W16, int C, int Cp, int O) {
    int total = O * Cp;
    sg_cvt_w<<<(total + 255) / 256, 256, 0, stream>>>(W, W16, C, Cp, total);
  };
  auto gemm = [&](const h16* W, const h16* X, const float* bias, float* Yf, h16* Yh,
                  int O, int rows, int Kp, int ldY, int colOff, float scale, int relu) {
    dim3 grid(O / 32, rows / 256), blk(32, 4);
    sg_gemm_wmma<<<grid, blk, 0, stream>>>(W, X, bias, Yf, Yh, Kp, ldY, colOff, scale, relu);
  };

  // ---- weight conversion ----
  for (int L = 0; L < 18; ++L) {
    int b0 = P_GNN + L * GS;
    cvt((const float*)d_in[b0 + 12], Wq[L], 256, 256, 256);
    cvt((const float*)d_in[b0 + 10], Wk[L], 256, 256, 256);
    cvt((const float*)d_in[b0 + 13], Wv[L], 256, 256, 256);
    cvt((const float*)d_in[b0 + 11], Wm[L], 256, 256, 256);
    cvt((const float*)d_in[b0 + 7],  W1[L], 512, 512, 512);
    cvt((const float*)d_in[b0 + 9],  W2[L], 512, 512, 256);
  }
  for (int i = 0; i < 5; ++i) {
    const float* wp = (i < 4) ? (const float*)d_in[P_KENC + 4 * i + 3]
                              : (const float*)d_in[P_KENC + 17];
    cvt(wp, KW[i], kC[i], kCp[i], kO[i]);
  }
  cvt(fw, FW16, 256, 256, 256);

  // ---- keypoint encoder ----
  sg_build_enc<<<(2048 * 32) / 256, 256, 0, stream>>>(kpts2d, sc2d, imsz, E16);
  const h16* cur = E16;
  for (int i = 0; i < 5; ++i) {
    const float* kb = (i < 4) ? (const float*)d_in[P_KENC + 4 * i]
                              : (const float*)d_in[P_KENC + 16];
    float* yf = (i == 4) ? kOut : kF;
    gemm(KW[i], cur, kb, yf, nullptr, kO[i], 2048, kCp[i], kO[i], 0, 1.f, 0);
    if (i < 4) {
      const float* kbt = (const float*)d_in[P_KENC + 4 * i + 1];
      const float* kg  = (const float*)d_in[P_KENC + 4 * i + 2];
      sg_bn_relu<<<kO[i], 256, 0, stream>>>(kF, kg, kbt, kA, 2048, kO[i]);
      cur = kA;
    }
  }
  sg_init_master<<<(2048 * 256) / 256, 256, 0, stream>>>(desc2d, kOut, dM, X16, 0);
  sg_init_master<<<(2048 * 256) / 256, 256, 0, stream>>>(desc3d, nullptr, dM, X16, 2048);

  // ---- GNN layers ----
  const int gtot = 16 * 1024 * 64;
  for (int L = 0; L < 18; ++L) {
    int b0 = P_GNN + L * GS;
    const float* bk  = (const float*)d_in[b0 + 0];
    const float* bm  = (const float*)d_in[b0 + 1];
    const float* bq  = (const float*)d_in[b0 + 2];
    const float* bv  = (const float*)d_in[b0 + 3];
    const float* m1b = (const float*)d_in[b0 + 4];
    const float* m1bt = (const float*)d_in[b0 + 5];
    const float* m1g = (const float*)d_in[b0 + 6];
    const float* m2b = (const float*)d_in[b0 + 8];
    bool cross = (L & 1) != 0;     // GNN_NAMES = ['self','cross']*9

    gemm(Wq[L], X16, bq, nullptr, Q16, 256, 4096, 256, 256, 0, 1.f, 0);
    if (!cross) {
      gemm(Wk[L], X16, bk, nullptr, K16, 256, 4096, 256, 256, 0, 1.f, 0);
      gemm(Wv[L], X16, bv, nullptr, V16, 256, 4096, 256, 256, 0, 1.f, 0);
    } else {  // half0 uses src=d1 (rows 2048..), half1 uses src=d0
      gemm(Wk[L], X16 + (size_t)2048 * 256, bk, nullptr, K16, 256, 2048, 256, 256, 0, 1.f, 0);
      gemm(Wk[L], X16, bk, nullptr, K16 + (size_t)2048 * 256, 256, 2048, 256, 256, 0, 1.f, 0);
      gemm(Wv[L], X16 + (size_t)2048 * 256, bv, nullptr, V16, 256, 2048, 256, 256, 0, 1.f, 0);
      gemm(Wv[L], X16, bv, nullptr, V16 + (size_t)2048 * 256, 256, 2048, 256, 256, 0, 1.f, 0);
    }
    sg_gather_nk<<<gtot / 256, 256, 0, stream>>>(Q16, Qh);
    sg_gather_nk<<<gtot / 256, 256, 0, stream>>>(K16, Kh);
    sg_gather_vT<<<gtot / 256, 256, 0, stream>>>(V16, VhT);
    for (int s = 0; s < 16; ++s) {
      const h16* q  = Qh + (size_t)s * 65536;
      const h16* k  = Kh + (size_t)s * 65536;
      const h16* vt = VhT + (size_t)s * 65536;
      h16* oh = Oh + (size_t)s * 65536;
      gemm(k, q, nullptr, Sc1, nullptr, 1024, 1024, 64, 1024, 0, 1.f, 0);
      sg_softmax_row<<<1024, 256, 0, stream>>>(Sc1, P16, 1024, 0.125f);  // 1/sqrt(64)
      gemm(vt, P16, nullptr, nullptr, oh, 64, 1024, 1024, 64, 0, 1.f, 0);
    }
    sg_scatter_heads<<<gtot / 256, 256, 0, stream>>>(Oh, Om16);
    // msg into cat[:,256:512]; x into cat[:,0:256]
    gemm(Wm[L], Om16, bm, nullptr, cat16, 256, 4096, 256, 512, 256, 1.f, 0);
    sg_copy_cols<<<(4096 * 256) / 256, 256, 0, stream>>>(X16, cat16);
    gemm(W1[L], cat16, m1b, m1f, nullptr, 512, 4096, 512, 512, 0, 1.f, 0);
    sg_bn_relu<<<512, 256, 0, stream>>>(m1f, m1g, m1bt, m1a, 2048, 512);
    sg_bn_relu<<<512, 256, 0, stream>>>(m1f + (size_t)2048 * 512, m1g, m1bt,
                                        m1a + (size_t)2048 * 512, 2048, 512);
    gemm(W2[L], m1a, m2b, m2f, nullptr, 256, 4096, 512, 256, 0, 1.f, 0);
    sg_residual<<<(4096 * 256) / 256, 256, 0, stream>>>(dM, m2f, X16, 4096 * 256);
  }

  // ---- final projection, scores, Sinkhorn, matching ----
  gemm(FW16, X16, fb, Mf, nullptr, 256, 4096, 256, 256, 0, 1.f, 0);
  sg_norm_rows<<<4096, 256, 0, stream>>>(Mf, M16);
  for (int b = 0; b < 2; ++b) {
    const h16* m1p = M16 + (size_t)(2048 + b * 1024) * 256;
    const h16* m0p = M16 + (size_t)(b * 1024) * 256;
    gemm(m1p, m0p, nullptr, Smatch + (size_t)b * 1024 * 1024, nullptr,
         1024, 1024, 256, 1024, 0, 1.f / 256.f, 0);
  }
  long cpt = 2L * 1025 * 1025;
  sg_build_cp<<<(int)((cpt + 255) / 256), 256, 0, stream>>>(Smatch, binsc, cp, cpt);
  sg_fill0<<<(2 * 1025 + 255) / 256, 256, 0, stream>>>(u, 2 * 1025);
  sg_fill0<<<(2 * 1025 + 255) / 256, 256, 0, stream>>>(v, 2 * 1025);
  float norm = -logf(2048.f);
  float lml = logf(1024.f) + norm;
  dim3 sgrid(1025, 2);
  for (int it = 0; it < 100; ++it) {
    sg_sink_u<<<sgrid, 256, 0, stream>>>(cp, v, u, norm, lml);
    sg_sink_v<<<sgrid, 256, 0, stream>>>(cp, u, v, norm, lml);
  }
  sg_conf<<<(int)((cpt + 255) / 256), 256, 0, stream>>>(cp, u, v, norm, cpt);
  dim3 mgrid(1024, 2);
  sg_rowmax<<<mgrid, 256, 0, stream>>>(cp, mx0, id0);
  sg_colmax<<<mgrid, 256, 0, stream>>>(cp, mx1, id1);
  sg_finalize<<<8, 256, 0, stream>>>(mx0, id0, mx1, id1, d_out);
}
